// FewShotSeg_57019985822411
// MI455X (gfx1250) — matrix operand order repaired and checked by reading, not compiled, not run on hardware
//
#include <hip/hip_runtime.h>

#define KARCH 5
#define MPK   5
#define TOPN  64
#define KNEI  16
#define NNODE 2048
#define NPIX  16384
#define CC    768
#define CR    192
#define LRELU 0.2f

typedef __attribute__((ext_vector_type(16))) _Float16 v16h;
typedef __attribute__((ext_vector_type(8)))  _Float16 v8h;
typedef __attribute__((ext_vector_type(8)))  float    v8f;
typedef unsigned long long u64;
typedef unsigned int u32;
typedef _Float16 f16;

// ============================ small utilities ============================
__global__ void k_f2h(const float* __restrict__ s, f16* __restrict__ d, int n) {
  int i = blockIdx.x * blockDim.x + threadIdx.x;
  if (i < n) d[i] = (f16)s[i];
}
__global__ void k_add(const float* __restrict__ a, const float* __restrict__ b,
                      float* __restrict__ c, int n) {
  int i = blockIdx.x * blockDim.x + threadIdx.x;
  if (i < n) c[i] = a[i] + b[i];
}
__global__ void k_zero32(u32* p, int n) {
  int i = blockIdx.x * blockDim.x + threadIdx.x;
  if (i < n) p[i] = 0u;
}

// f32 [NNODE, CC] -> f16 [CC, NNODE] tiled transpose (feeds the nt GEMM).
__global__ void k_transpose_h(const float* __restrict__ src, f16* __restrict__ dst) {
  __shared__ float tile[32][33];
  int bx = blockIdx.x, by = blockIdx.y;            // bx: CC/32, by: NNODE/32
  int tx = threadIdx.x & 31, ty = threadIdx.x >> 5; // 32 x 8
  for (int r = 0; r < 32; r += 8)
    tile[ty + r][tx] = src[(size_t)(by * 32 + ty + r) * CC + bx * 32 + tx];
  __syncthreads();
  for (int r = 0; r < 32; r += 8)
    dst[(size_t)(bx * 32 + ty + r) * NNODE + by * 32 + tx] = (f16)tile[tx][ty + r];
}

// ============================ FPS ============================
// feat layout: [C, NPIX] channel-major (as given) -> coalesced per-pixel access.
__global__ void k_norms(const float* __restrict__ feat, float* __restrict__ n2) {
  int p = blockIdx.x * 256 + threadIdx.x;
  float s = 0.f;
  for (int c = 0; c < CC; ++c) { float v = feat[c * NPIX + p]; s += v * v; }
  n2[p] = s;
}

__global__ void k_fps_init(float* dist, u64* packed) {
  int p = blockIdx.x * 256 + threadIdx.x;
  dist[p] = 1e10f;
  if (blockIdx.x == 0 && threadIdx.x < 3) packed[threadIdx.x] = 0ull;
}

// One step of FPS. Reads winner of step t-1 from packed[(t+2)%3], accumulates
// new argmax into packed[t%3], resets packed[(t+1)%3] for the next step.
__global__ void k_fps_step(const float* __restrict__ feat,
                           const float* __restrict__ n2,
                           float* __restrict__ dist,
                           u64* __restrict__ packed,
                           int* __restrict__ fps_idx, int t) {
  __shared__ float farv[CC];
  __shared__ u64 red[256];
  int tid = threadIdx.x;
  int p = blockIdx.x * 256 + tid;

  int far = 0;
  if (t > 0) far = NPIX - 1 - (int)(u32)(packed[(t + 2) % 3] & 0xFFFFFFFFull);

  for (int c = tid; c < CC; c += 256) farv[c] = feat[c * NPIX + far];
  __syncthreads();

  float dot = 0.f;
  for (int c = 0; c < CC; ++c) dot += feat[c * NPIX + p] * farv[c];
  float d = n2[p] + n2[far] - 2.f * dot;
  float nd = fminf(dist[p], d);
  dist[p] = nd;

  // nd >= 0 -> raw float bits are order-preserving; encode NPIX-1-p so ties
  // pick the smallest index (matches jnp.argmax).
  u64 pk = ((u64)__float_as_uint(nd) << 32) | (u32)(NPIX - 1 - p);
  red[tid] = pk;
  __syncthreads();
  for (int s = 128; s > 0; s >>= 1) {
    if (tid < s) red[tid] = red[tid] > red[tid + s] ? red[tid] : red[tid + s];
    __syncthreads();
  }
  if (tid == 0) {
    atomicMax(&packed[t % 3], red[0]);
    packed[(t + 1) % 3] = 0ull;           // same value from all blocks: benign
    if (blockIdx.x == 0) fps_idx[t] = far;
  }
}

// Gather selected nodes; emit g (f32), g16, gn16 (l2-normalized f16).
__global__ void k_gather(const float* __restrict__ feat,
                         const int* __restrict__ fps_idx,
                         float* __restrict__ g, f16* __restrict__ g16,
                         f16* __restrict__ gn16) {
  __shared__ float row[CC];
  __shared__ float red[256];
  int i = blockIdx.x, tid = threadIdx.x;
  int idx = fps_idx[i];
  float ss = 0.f;
  for (int c = tid; c < CC; c += 256) {
    float v = feat[c * NPIX + idx];
    row[c] = v;
    g[i * CC + c] = v;
    g16[i * CC + c] = (f16)v;
    ss += v * v;
  }
  red[tid] = ss;
  __syncthreads();
  for (int s = 128; s > 0; s >>= 1) {
    if (tid < s) red[tid] += red[tid + s];
    __syncthreads();
  }
  float nrm = fmaxf(sqrtf(red[0]), 1e-12f);
  for (int c = tid; c < CC; c += 256) gn16[i * CC + c] = (f16)(row[c] / nrm);
}

// ============================ WMMA GEMM (nt, direct-global fragments) ======
// C[M,N] = A[M,K] * B[N,K]^T, A/B row-major f16, C f32.
// One wave computes a 16(M) x 64(N) slab: A fragment loaded once per K-step,
// reused by 4 back-to-back WMMAs. Every lane's fragment slice is contiguous
// in memory (two aligned 16B loads), using a contiguous k order shared by A and B:
//   A (16x32): lanes 0-15 row m hold K 0-15; lanes 16-31 hold K 16-31
//   B (32x16): lanes 0-15 col n hold K 0-15; lanes 16-31 hold K 16-31
//   D (16x16): lane l reg r -> row r + 8*(l>=16), col l&15
__global__ void __launch_bounds__(32)
k_gemm_nt(const f16* __restrict__ A, const f16* __restrict__ B,
          float* __restrict__ C, int M, int N, int K) {
  int tm = blockIdx.x;              // M / 16
  int tn4 = blockIdx.y;             // N / 64
  int l = threadIdx.x;
  int lane16 = l & 15, hi = l >> 4;

  const f16* arow = A + (size_t)(tm * 16 + lane16) * K;
  const f16* b0r  = B + (size_t)(tn4 * 64 +  0 + lane16) * K;
  const f16* b1r  = B + (size_t)(tn4 * 64 + 16 + lane16) * K;
  const f16* b2r  = B + (size_t)(tn4 * 64 + 32 + lane16) * K;
  const f16* b3r  = B + (size_t)(tn4 * 64 + 48 + lane16) * K;

  v8f acc0 = {}, acc1 = {}, acc2 = {}, acc3 = {};
  for (int kb = 0; kb < K; kb += 32) {
    v8h alo = *(const v8h*)(arow + kb + hi * 16);
    v8h ahi = *(const v8h*)(arow + kb + hi * 16 + 8);
    v16h a = __builtin_shufflevector(alo, ahi, 0, 1, 2, 3, 4, 5, 6, 7,
                                     8, 9, 10, 11, 12, 13, 14, 15);
    int ko = kb + hi * 16;
    v8h p, q;
    p = *(const v8h*)(b0r + ko); q = *(const v8h*)(b0r + ko + 8);
    v16h b0 = __builtin_shufflevector(p, q, 0,1,2,3,4,5,6,7,8,9,10,11,12,13,14,15);
    p = *(const v8h*)(b1r + ko); q = *(const v8h*)(b1r + ko + 8);
    v16h b1 = __builtin_shufflevector(p, q, 0,1,2,3,4,5,6,7,8,9,10,11,12,13,14,15);
    p = *(const v8h*)(b2r + ko); q = *(const v8h*)(b2r + ko + 8);
    v16h b2 = __builtin_shufflevector(p, q, 0,1,2,3,4,5,6,7,8,9,10,11,12,13,14,15);
    p = *(const v8h*)(b3r + ko); q = *(const v8h*)(b3r + ko + 8);
    v16h b3 = __builtin_shufflevector(p, q, 0,1,2,3,4,5,6,7,8,9,10,11,12,13,14,15);

    acc0 = __builtin_amdgcn_wmma_f32_16x16x32_f16(false, a, false, b0, (short)0, acc0, false, false);
    acc1 = __builtin_amdgcn_wmma_f32_16x16x32_f16(false, a, false, b1, (short)0, acc1, false, false);
    acc2 = __builtin_amdgcn_wmma_f32_16x16x32_f16(false, a, false, b2, (short)0, acc2, false, false);
    acc3 = __builtin_amdgcn_wmma_f32_16x16x32_f16(false, a, false, b3, (short)0, acc3, false, false);
  }

  int col = lane16, rbase = hi * 8;
  for (int r = 0; r < 8; ++r) {
    float* crow = C + (size_t)(tm * 16 + rbase + r) * N + tn4 * 64 + col;
    crow[0]  = acc0[r];
    crow[16] = acc1[r];
    crow[32] = acc2[r];
    crow[48] = acc3[r];
  }
}

// ============================ GAT pieces ============================
__global__ void k_elr(const float* __restrict__ Wh, const float* __restrict__ a_gat,
                      float* __restrict__ el, float* __restrict__ er) {
  __shared__ float r0[256], r1[256];
  int i = blockIdx.x, t = threadIdx.x;
  float s0 = 0.f, s1 = 0.f;
  for (int c = t; c < CC; c += 256) {
    float w = Wh[i * CC + c];
    s0 += w * a_gat[c];
    s1 += w * a_gat[CC + c];
  }
  r0[t] = s0; r1[t] = s1;
  __syncthreads();
  for (int s = 128; s > 0; s >>= 1) {
    if (t < s) { r0[t] += r0[t + s]; r1[t] += r1[t + s]; }
    __syncthreads();
  }
  if (t == 0) { el[i] = r0[0]; er[i] = r1[0]; }
}

__global__ void k_topk16(const float* __restrict__ sim, unsigned char* __restrict__ mask) {
  __shared__ float vals[NNODE];
  __shared__ u64 red[256];
  int i = blockIdx.x, t = threadIdx.x;
  for (int j = t; j < NNODE; j += 256) vals[j] = sim[(size_t)i * NNODE + j];
  __syncthreads();
  for (int it = 0; it < KNEI; ++it) {
    u64 best = 0;
    for (int j = t; j < NNODE; j += 256) {
      u32 bits = __float_as_uint(vals[j]);
      bits ^= (bits & 0x80000000u) ? 0xFFFFFFFFu : 0x80000000u;  // order-preserving map
      u64 pk = ((u64)bits << 32) | (u32)(NNODE - 1 - j);
      best = pk > best ? pk : best;
    }
    red[t] = best;
    __syncthreads();
    for (int s = 128; s > 0; s >>= 1) {
      if (t < s) red[t] = red[t] > red[t + s] ? red[t] : red[t + s];
      __syncthreads();
    }
    if (t == 0) {
      int j = NNODE - 1 - (int)(u32)(red[0] & 0xFFFFFFFFull);
      vals[j] = -3.0e38f;
      mask[(size_t)i * NNODE + j] = 1;
      mask[(size_t)j * NNODE + i] = 1;
    }
    __syncthreads();
  }
}

// att row: v = mask ? ((sim>0 ? lrelu(el+er) : -9e15) * sim) : 0 ; row softmax -> f16
__global__ void k_att(const float* __restrict__ sim, const unsigned char* __restrict__ mask,
                      const float* __restrict__ el, const float* __restrict__ er,
                      f16* __restrict__ att) {
  __shared__ float vals[NNODE];
  __shared__ float red[256];
  int i = blockIdx.x, t = threadIdx.x;
  float eli = el[i];
  float mx = -3.0e38f;
  for (int j = t; j < NNODE; j += 256) {
    float v = 0.f;
    if (mask[(size_t)i * NNODE + j]) {
      float s = sim[(size_t)i * NNODE + j];
      float e = eli + er[j];
      e = e > 0.f ? e : LRELU * e;
      v = (s > 0.f ? e : -9.0e15f) * s;
    }
    vals[j] = v;
    mx = fmaxf(mx, v);
  }
  red[t] = mx;
  __syncthreads();
  for (int s = 128; s > 0; s >>= 1) { if (t < s) red[t] = fmaxf(red[t], red[t + s]); __syncthreads(); }
  mx = red[0];
  __syncthreads();
  float sum = 0.f;
  for (int j = t; j < NNODE; j += 256) sum += __expf(vals[j] - mx);
  red[t] = sum;
  __syncthreads();
  for (int s = 128; s > 0; s >>= 1) { if (t < s) red[t] += red[t + s]; __syncthreads(); }
  float inv = 1.0f / red[0];
  for (int j = t; j < NNODE; j += 256)
    att[(size_t)i * NNODE + j] = (f16)(__expf(vals[j] - mx) * inv);
}

// ============================ SE refine ============================
__global__ void k_colmean(const float* __restrict__ x, float* __restrict__ meanv) {
  int c = blockIdx.x * 256 + threadIdx.x;     // 3 blocks * 256 = 768
  float s = 0.f;
  for (int n = 0; n < NNODE; ++n) s += x[(size_t)n * CC + c];
  meanv[c] = s / (float)NNODE;
}

__global__ void k_se(const float* __restrict__ meanv, const float* __restrict__ w1,
                     const float* __restrict__ w2, float* __restrict__ yv) {
  __shared__ float mv[CC];
  __shared__ float h1[CR];
  int t = threadIdx.x;
  for (int c = t; c < CC; c += 256) mv[c] = meanv[c];
  __syncthreads();
  for (int j = t; j < CR; j += 256) {
    float s = 0.f;
    for (int c = 0; c < CC; ++c) s += mv[c] * w1[j * CC + c];
    h1[j] = s > 0.f ? s : 0.f;
  }
  __syncthreads();
  for (int c = t; c < CC; c += 256) {
    float s = 0.f;
    for (int j = 0; j < CR; ++j) s += h1[j] * w2[c * CR + j];
    yv[c] = 1.0f / (1.0f + __expf(-s));
  }
}

__global__ void k_refined(const float* __restrict__ x, const float* __restrict__ yv,
                          float* __restrict__ refined, float* __restrict__ rn) {
  __shared__ float row[CC];
  __shared__ float red[256];
  int n = blockIdx.x, t = threadIdx.x;
  float ss = 0.f;
  for (int c = t; c < CC; c += 256) {
    float v = x[(size_t)n * CC + c] * yv[c];
    row[c] = v;
    refined[(size_t)n * CC + c] = v;
    ss += v * v;
  }
  red[t] = ss;
  __syncthreads();
  for (int s = 128; s > 0; s >>= 1) { if (t < s) red[t] += red[t + s]; __syncthreads(); }
  float nrm = fmaxf(sqrtf(red[0]), 1e-12f);
  for (int c = t; c < CC; c += 256) rn[(size_t)n * CC + c] = row[c] / nrm;
}

// ============================ prototypes ============================
__global__ void k_canon(const float* __restrict__ cq, const float* __restrict__ rn,
                        const float* __restrict__ refined, float* __restrict__ aff,
                        float* __restrict__ out) {
  __shared__ float q[CC];
  __shared__ float red[256];
  __shared__ float w[NNODE];
  int k = blockIdx.x, t = threadIdx.x;
  float ss = 0.f;
  for (int c = t; c < CC; c += 256) { float v = cq[k * CC + c]; q[c] = v; ss += v * v; }
  red[t] = ss;
  __syncthreads();
  for (int s = 128; s > 0; s >>= 1) { if (t < s) red[t] += red[t + s]; __syncthreads(); }
  float nrm = fmaxf(sqrtf(red[0]), 1e-12f);
  __syncthreads();

  float mx = -3.0e38f;
  for (int n = t; n < NNODE; n += 256) {
    float d = 0.f;
    for (int c = 0; c < CC; ++c) d += q[c] * rn[(size_t)n * CC + c];
    d /= nrm;
    aff[k * NNODE + n] = d;
    w[n] = d;
    mx = fmaxf(mx, d);
  }
  red[t] = mx;
  __syncthreads();
  for (int s = 128; s > 0; s >>= 1) { if (t < s) red[t] = fmaxf(red[t], red[t + s]); __syncthreads(); }
  mx = red[0];
  __syncthreads();
  float sum = 0.f;
  for (int n = t; n < NNODE; n += 256) { float e = __expf(w[n] - mx); w[n] = e; sum += e; }
  red[t] = sum;
  __syncthreads();
  for (int s = 128; s > 0; s >>= 1) { if (t < s) red[t] += red[t + s]; __syncthreads(); }
  float inv = 1.0f / red[0];
  __syncthreads();
  for (int n = t; n < NNODE; n += 256) w[n] *= inv;
  __syncthreads();

  for (int c = t; c < CC; c += 256) {
    float acc = 0.f;
    for (int n = 0; n < NNODE; ++n) acc += w[n] * refined[(size_t)n * CC + c];
    out[k * CC + c] = acc;   // rows 0..4 of output
  }
}

__global__ void k_top64(const float* __restrict__ aff, int* __restrict__ topidx) {
  __shared__ float vals[NNODE];
  __shared__ u64 red[256];
  int k = blockIdx.x, t = threadIdx.x;
  for (int j = t; j < NNODE; j += 256) vals[j] = aff[k * NNODE + j];
  __syncthreads();
  for (int it = 0; it < TOPN; ++it) {
    u64 best = 0;
    for (int j = t; j < NNODE; j += 256) {
      u32 bits = __float_as_uint(vals[j]);
      bits ^= (bits & 0x80000000u) ? 0xFFFFFFFFu : 0x80000000u;
      u64 pk = ((u64)bits << 32) | (u32)(NNODE - 1 - j);
      best = pk > best ? pk : best;
    }
    red[t] = best;
    __syncthreads();
    for (int s = 128; s > 0; s >>= 1) {
      if (t < s) red[t] = red[t] > red[t + s] ? red[t] : red[t + s];
      __syncthreads();
    }
    if (t == 0) {
      int j = NNODE - 1 - (int)(u32)(red[0] & 0xFFFFFFFFull);
      vals[j] = -3.0e38f;
      topidx[k * TOPN + it] = j;
    }
    __syncthreads();
  }
}

__global__ void k_app(const float* __restrict__ aq, const float* __restrict__ rn,
                      const float* __restrict__ refined, const int* __restrict__ topidx,
                      float* __restrict__ out) {
  __shared__ float q[CC];
  __shared__ float red[256];
  __shared__ float sv[TOPN];
  __shared__ int idxs[TOPN];
  int k = blockIdx.x / MPK, m = blockIdx.x % MPK, t = threadIdx.x;
  float ss = 0.f;
  for (int c = t; c < CC; c += 256) { float v = aq[m * CC + c]; q[c] = v; ss += v * v; }
  red[t] = ss;
  if (t < TOPN) idxs[t] = topidx[k * TOPN + t];
  __syncthreads();
  for (int s = 128; s > 0; s >>= 1) { if (t < s) red[t] += red[t + s]; __syncthreads(); }
  float nrm = fmaxf(sqrtf(red[0]), 1e-12f);
  __syncthreads();
  for (int c = t; c < CC; c += 256) q[c] /= nrm;
  __syncthreads();

  if (t < TOPN) {
    float d = 0.f;
    int idx = idxs[t];
    for (int c = 0; c < CC; ++c) d += q[c] * rn[(size_t)idx * CC + c];
    sv[t] = d;
  }
  __syncthreads();
  if (t == 0) {
    float mx = -3.0e38f;
    for (int n = 0; n < TOPN; ++n) mx = fmaxf(mx, sv[n]);
    float sum = 0.f;
    for (int n = 0; n < TOPN; ++n) { float e = __expf(sv[n] - mx); sv[n] = e; sum += e; }
    float inv = 1.0f / sum;
    for (int n = 0; n < TOPN; ++n) sv[n] *= inv;
  }
  __syncthreads();
  for (int c = t; c < CC; c += 256) {
    float acc = 0.f;
    for (int n = 0; n < TOPN; ++n) acc += sv[n] * refined[(size_t)idxs[n] * CC + c];
    out[(KARCH + k * MPK + m) * CC + c] = acc;  // rows 5..29 of output
  }
}

// ============================ host launcher ============================
extern "C" void kernel_launch(void* const* d_in, const int* in_sizes, int n_in,
                              void* d_out, int out_size, void* d_ws, size_t ws_size,
                              hipStream_t stream) {
  (void)in_sizes; (void)n_in; (void)out_size; (void)ws_size;
  const float* feat = (const float*)d_in[0];   // [1,768,128,128] -> [C, NPIX]
  const float* cq   = (const float*)d_in[3];   // [5,768]
  const float* aq   = (const float*)d_in[4];   // [5,768]
  const float* Wg   = (const float*)d_in[5];   // [768,768]
  const float* ag   = (const float*)d_in[6];   // [1,1536]
  const float* sw1  = (const float*)d_in[7];   // [192,768]
  const float* sw2  = (const float*)d_in[8];   // [768,192]
  float* out = (float*)d_out;

  char* ws = (char*)d_ws;
  size_t off = 0;
  auto alloc = [&](size_t bytes) -> void* {
    void* p = ws + off;
    off += (bytes + 255) & ~(size_t)255;
    return p;
  };
  float* n2     = (float*)alloc(NPIX * 4);
  float* dist   = (float*)alloc(NPIX * 4);
  u64*   packed = (u64*)  alloc(3 * 8);
  int*   fpsidx = (int*)  alloc(NNODE * 4);
  float* g      = (float*)alloc((size_t)NNODE * CC * 4);
  f16*   g16    = (f16*)  alloc((size_t)NNODE * CC * 2);
  f16*   gn16   = (f16*)  alloc((size_t)NNODE * CC * 2);
  f16*   W16    = (f16*)  alloc((size_t)CC * CC * 2);
  float* Wh     = (float*)alloc((size_t)NNODE * CC * 4);
  f16*   WhT16  = (f16*)  alloc((size_t)CC * NNODE * 2);   // Wh transposed, f16
  float* el     = (float*)alloc(NNODE * 4);
  float* er     = (float*)alloc(NNODE * 4);
  float* sim    = (float*)alloc((size_t)NNODE * NNODE * 4);
  unsigned char* mask = (unsigned char*)alloc((size_t)NNODE * NNODE);
  f16*   att16  = (f16*)  alloc((size_t)NNODE * NNODE * 2);
  float* hp     = (float*)alloc((size_t)NNODE * CC * 4);
  float* x      = (float*)alloc((size_t)NNODE * CC * 4);
  float* meanv  = (float*)alloc(CC * 4);
  float* yv     = (float*)alloc(CC * 4);
  float* refined= (float*)alloc((size_t)NNODE * CC * 4);
  float* rn     = (float*)alloc((size_t)NNODE * CC * 4);
  float* aff    = (float*)alloc((size_t)KARCH * NNODE * 4);
  int*   topidx = (int*)  alloc(KARCH * TOPN * 4);

  // ---- FPS (sequential chain of L2-resident GEMV+argmax steps) ----
  k_norms<<<NPIX / 256, 256, 0, stream>>>(feat, n2);
  k_fps_init<<<NPIX / 256, 256, 0, stream>>>(dist, packed);
  for (int t = 0; t < NNODE; ++t)
    k_fps_step<<<NPIX / 256, 256, 0, stream>>>(feat, n2, dist, packed, fpsidx, t);
  k_gather<<<NNODE, 256, 0, stream>>>(feat, fpsidx, g, g16, gn16);

  // ---- GAT: Wh = g @ W^T, sim = gn @ gn^T (WMMA f16 -> f32, nt form) ----
  k_f2h<<<(CC * CC + 255) / 256, 256, 0, stream>>>(Wg, W16, CC * CC);
  k_gemm_nt<<<dim3(NNODE / 16, CC / 64), 32, 0, stream>>>(g16, W16, Wh, NNODE, CC, CC);
  k_transpose_h<<<dim3(CC / 32, NNODE / 32), 256, 0, stream>>>(Wh, WhT16);
  k_elr<<<NNODE, 256, 0, stream>>>(Wh, ag, el, er);
  k_gemm_nt<<<dim3(NNODE / 16, NNODE / 64), 32, 0, stream>>>(gn16, gn16, sim, NNODE, NNODE, CC);

  // ---- kNN adjacency + masked attention softmax ----
  k_zero32<<<(NNODE * NNODE / 4 + 255) / 256, 256, 0, stream>>>((u32*)mask, NNODE * NNODE / 4);
  k_topk16<<<NNODE, 256, 0, stream>>>(sim, mask);
  k_att<<<NNODE, 256, 0, stream>>>(sim, mask, el, er, att16);

  // ---- h' = att @ Wh = att @ (WhT)^T (WMMA nt), x = g + h' ----
  k_gemm_nt<<<dim3(NNODE / 16, CC / 64), 32, 0, stream>>>(att16, WhT16, hp, NNODE, CC, NNODE);
  k_add<<<(NNODE * CC + 255) / 256, 256, 0, stream>>>(g, hp, x, NNODE * CC);

  // ---- SE refine ----
  k_colmean<<<CC / 256, 256, 0, stream>>>(x, meanv);
  k_se<<<1, 256, 0, stream>>>(meanv, sw1, sw2, yv);
  k_refined<<<NNODE, 256, 0, stream>>>(x, yv, refined, rn);

  // ---- canonical + appearance prototypes (written straight into d_out) ----
  k_canon<<<KARCH, 256, 0, stream>>>(cq, rn, refined, aff, out);
  k_top64<<<KARCH, 256, 0, stream>>>(aff, topidx);
  k_app<<<KARCH * MPK, 256, 0, stream>>>(aq, rn, refined, topidx, out);
}
